// my_param_layer_60842506715779
// MI455X (gfx1250) — compile-verified, matrix operand order checked by reading, not run on hardware
//
#include <hip/hip_runtime.h>

// ---------------------------------------------------------------------------
// FP32 4096^3 GEMM via bf16x3 split on CDNA5 WMMA (gfx1250, wave32).
//   C = A*B, fp32 in/out.  x = hi + lo (hardware-RNE bf16 split);
//   C ~= Ah*Bh + Ah*Bl + Al*Bh with f32 WMMA accumulation.
// Staging uses native f32<->bf16 converts (v_cvt_*bf16*) and pure b128 LDS
// traffic: 8x ds_store_b128 per thread/iter, b128 fragment loads.
// ---------------------------------------------------------------------------

#define M_DIM 4096
#define N_DIM 4096
#define K_DIM 4096

constexpr int BM  = 128;   // C tile rows per workgroup
constexpr int BN  = 128;   // C tile cols per workgroup
constexpr int BK  = 32;    // k-chunk (== WMMA bf16 K)
constexpr int BKP = 40;    // padded LDS row stride (80B: 16B aligned, bank-skewed)
constexpr int NK  = K_DIM / BK;

typedef __attribute__((ext_vector_type(16))) __bf16        v16bf;
typedef __attribute__((ext_vector_type(8)))  __bf16        v8bf;   // 16 bytes
typedef __attribute__((ext_vector_type(8)))  float         v8f;
typedef __attribute__((ext_vector_type(4)))  unsigned int  v4u;

__global__ __launch_bounds__(256)
void gemm_f32_bf16x3(const float* __restrict__ A,
                     const float* __restrict__ B,
                     float* __restrict__ C) {
    __shared__ __align__(16) unsigned short sAh[2][BM][BKP];
    __shared__ __align__(16) unsigned short sAl[2][BM][BKP];
    __shared__ __align__(16) unsigned short sBh[2][BN][BKP];  // transposed: [n][k]
    __shared__ __align__(16) unsigned short sBl[2][BN][BKP];

    const int tid  = threadIdx.x;
    const int lane = tid & 31;
    const int wave = tid >> 5;
    const int wm   = wave & 3;               // 0..3 : 32-row slab of C tile
    const int wn   = wave >> 2;              // 0..1 : 64-col slab of C tile
    const int lr   = lane & 15;              // row/col within 16x16 fragment
    const int lh   = lane >> 4;              // half-select
    const int ksel = lh << 3;                // 0 or 8 (16-bit operand layout)

    const size_t bm = (size_t)blockIdx.y * BM;
    const size_t bn = (size_t)blockIdx.x * BN;

    // Staging ownership: thread -> one tile row (A) / one tile col (B),
    // covering 16 contiguous k-values (half of BK).
    const int srow = tid & 127;              // A row / B column within tile
    const int sk16 = (tid >> 7) << 4;        // 0 or 16: k-half

    const float* aGbase = A + (bm + srow) * (size_t)K_DIM + sk16;
    const float* bGbase = B + (size_t)sk16 * N_DIM + bn + srow;

    float aF[16], bF[16];

    auto loadGlobal = [&](int k0) {
        #pragma unroll
        for (int q = 0; q < 4; ++q)          // 4x coalesced b128 along k
            *(float4*)&aF[q * 4] = *(const float4*)(aGbase + k0 + q * 4);
        #pragma unroll
        for (int j = 0; j < 16; ++j)         // k-strided scalars, lane-coalesced in n
            bF[j] = bGbase[(size_t)(k0 + j) * N_DIM];
    };

    auto prefetchGlobal = [&](int k0) {
        __builtin_prefetch(aGbase + k0, 0, 1);
        __builtin_prefetch(bGbase + (size_t)k0 * N_DIM, 0, 1);
    };

    // Hardware bf16 split of 8 floats -> hi/lo vectors (one b128 store each).
    auto split8 = [](const float* src, v8bf& hv, v8bf& lv) {
        #pragma unroll
        for (int e = 0; e < 8; ++e) {
            __bf16 h = (__bf16)src[e];            // v_cvt(_pk)_bf16_f32, RNE
            hv[e] = h;
            lv[e] = (__bf16)(src[e] - (float)h);  // exact residual, RNE to bf16
        }
    };

    auto stageLds = [&](int buf) {
        v8bf hv, lv;
        #pragma unroll
        for (int q = 0; q < 2; ++q) {
            split8(&aF[q * 8], hv, lv);
            *(v8bf*)&sAh[buf][srow][sk16 + q * 8] = hv;
            *(v8bf*)&sAl[buf][srow][sk16 + q * 8] = lv;
        }
        #pragma unroll
        for (int q = 0; q < 2; ++q) {
            split8(&bF[q * 8], hv, lv);
            *(v8bf*)&sBh[buf][srow][sk16 + q * 8] = hv;   // transposed for free
            *(v8bf*)&sBl[buf][srow][sk16 + q * 8] = lv;
        }
    };

    // 16-bit operand fragment: lane holds row (A) / col (B) = lr, k-values
    // ksel..ksel+7 and ksel+16..ksel+23 -> two b128 LDS loads.
    auto frag = [&](const unsigned short (*src)[BKP], int row) -> v16bf {
        union { v4u q[2]; v16bf f; } u;
        u.q[0] = *(const v4u*)&src[row][ksel];
        u.q[1] = *(const v4u*)&src[row][ksel + 16];
        return u.f;
    };

    v8f acc[2][4];
    #pragma unroll
    for (int t = 0; t < 2; ++t)
        #pragma unroll
        for (int n = 0; n < 4; ++n)
            acc[t][n] = (v8f){0.f, 0.f, 0.f, 0.f, 0.f, 0.f, 0.f, 0.f};

    loadGlobal(0);
    int buf = 0;

    for (int kt = 0; kt < NK; ++kt) {
        stageLds(buf);
        __syncthreads();
        if (kt + 1 < NK) loadGlobal((kt + 1) * BK);       // register prefetch (+1)
        if (kt + 2 < NK) prefetchGlobal((kt + 2) * BK);   // L2 prefetch (+2)

        v16bf aHi[2], aLo[2];
        #pragma unroll
        for (int t = 0; t < 2; ++t) {
            aHi[t] = frag(sAh[buf], wm * 32 + t * 16 + lr);
            aLo[t] = frag(sAl[buf], wm * 32 + t * 16 + lr);
        }
        #pragma unroll
        for (int n = 0; n < 4; ++n) {
            v16bf bHi = frag(sBh[buf], wn * 64 + n * 16 + lr);
            v16bf bLo = frag(sBl[buf], wn * 64 + n * 16 + lr);
            #pragma unroll
            for (int t = 0; t < 2; ++t) {
                acc[t][n] = __builtin_amdgcn_wmma_f32_16x16x32_bf16(
                    false, aHi[t], false, bHi, (short)0, acc[t][n], false, false);
                acc[t][n] = __builtin_amdgcn_wmma_f32_16x16x32_bf16(
                    false, aHi[t], false, bLo, (short)0, acc[t][n], false, false);
                acc[t][n] = __builtin_amdgcn_wmma_f32_16x16x32_bf16(
                    false, aLo[t], false, bHi, (short)0, acc[t][n], false, false);
            }
        }
        buf ^= 1;
        // One barrier per iteration suffices: the next stage writes the other
        // buffer; this buffer can only be re-written after every wave passed
        // the *next* iteration's barrier (i.e., finished reading it).
    }

    // Epilogue: C/D layout — VGPR i: lanes 0-15 -> M=i, lanes 16-31 -> M=8+i; N=lane%16.
    #pragma unroll
    for (int t = 0; t < 2; ++t) {
        #pragma unroll
        for (int n = 0; n < 4; ++n) {
            #pragma unroll
            for (int i = 0; i < 8; ++i) {
                size_t row = bm + wm * 32 + t * 16 + i + lh * 8;
                size_t col = bn + wn * 64 + n * 16 + lr;
                C[row * (size_t)N_DIM + col] = acc[t][n][i];
            }
        }
    }
}

extern "C" void kernel_launch(void* const* d_in, const int* in_sizes, int n_in,
                              void* d_out, int out_size, void* d_ws, size_t ws_size,
                              hipStream_t stream) {
    const float* A = (const float*)d_in[0];
    const float* B = (const float*)d_in[1];
    float*       C = (float*)d_out;
    dim3 grid(N_DIM / BN, M_DIM / BM);   // 32 x 32 workgroups
    gemm_f32_bf16x3<<<grid, 256, 0, stream>>>(A, B, C);
}
